// GPT2Attention_63264868270313
// MI455X (gfx1250) — compile-verified, matrix operand order checked by reading, not run on hardware
//
#include <hip/hip_runtime.h>

typedef __attribute__((ext_vector_type(16))) _Float16 v16h;
typedef __attribute__((ext_vector_type(8)))  _Float16 v8h;
typedef __attribute__((ext_vector_type(8)))  float    v8f;

static constexpr int Tt  = 2048;
static constexpr int Cc  = 2048;
static constexpr int Hn  = 16;
static constexpr int HDd = 128;

union Frag16 { v16h h; uint4 u[2]; };
union H8     { v8h  h; uint4 u;    };

__device__ inline uint4 pack8(float4 a, float4 b) {
  H8 t;
  t.h[0] = (_Float16)a.x; t.h[1] = (_Float16)a.y;
  t.h[2] = (_Float16)a.z; t.h[3] = (_Float16)a.w;
  t.h[4] = (_Float16)b.x; t.h[5] = (_Float16)b.y;
  t.h[6] = (_Float16)b.z; t.h[7] = (_Float16)b.w;
  return t.u;
}

__device__ inline v8f wmma16(const Frag16& a, const Frag16& b, v8f c) {
  // D = A(16x32 f16) * B(32x16 f16) + C(16x16 f32)
  return __builtin_amdgcn_wmma_f32_16x16x32_f16(
      /*neg_a=*/false, a.h, /*neg_b=*/false, b.h,
      /*c_mod=*/(short)0, c, /*reuse_a=*/false, /*reuse_b=*/false);
}

// ---- DPP16 butterfly reductions across a 16-lane row group ---------------
template <int CTRL>
__device__ inline float dpp_xor(float x) {
  int v = __builtin_amdgcn_update_dpp(0, __float_as_int(x), CTRL, 0xf, 0xf, true);
  return __int_as_float(v);
}
__device__ inline float rowmax16(float x) {
  x = fmaxf(x, dpp_xor<0xB1>(x));   // quad_perm(1,0,3,2)
  x = fmaxf(x, dpp_xor<0x4E>(x));   // quad_perm(2,3,0,1)
  x = fmaxf(x, dpp_xor<0x141>(x));  // row_half_mirror
  x = fmaxf(x, dpp_xor<0x140>(x));  // row_mirror
  return x;
}
__device__ inline float rowsum16(float x) {
  x += dpp_xor<0xB1>(x);
  x += dpp_xor<0x4E>(x);
  x += dpp_xor<0x141>(x);
  x += dpp_xor<0x140>(x);
  return x;
}

// ---------------------------------------------------------------------------
// One-time f32 -> f16 conversion (weights / once-read tensors). 8 elems/thread.
// ---------------------------------------------------------------------------
__global__ __launch_bounds__(256) void cvt_f32_f16_kernel(
    const float* __restrict__ src, _Float16* __restrict__ dst, int n8) {
  int i = blockIdx.x * 256 + threadIdx.x;
  if (i < n8) {
    const float* p = src + (size_t)i * 8;
    float4 a = *(const float4*)p;
    float4 b = *(const float4*)(p + 4);
    *(uint4*)(dst + (size_t)i * 8) = pack8(a, b);
  }
}

// ---------------------------------------------------------------------------
// GEMM: out = A[M,K] * W[N,K]^T + bias[N]   (W already f16)
// grid.x = N/128, grid.y = M/64, block = 256 (8 waves).
// Block tile 64x128; each wave computes a 64x16 strip (4 accumulators).
// EPI==0: A is f32; epilogue scatters q (f16, pre-scaled) / k,v (f32).
// EPI==1: A is f16 (staged via global_load_async_to_lds); plain f32 store.
// ---------------------------------------------------------------------------
template <int EPI>
__global__ __launch_bounds__(256) void gemm_wmma_kernel(
    const void* __restrict__ Av, const _Float16* __restrict__ W,
    const float* __restrict__ bias, void* __restrict__ out0,
    float* __restrict__ out1, float* __restrict__ out2, int K) {
  __shared__ _Float16 Alds[64 * 32];  // 4 KB

  const int tid  = threadIdx.x;
  const int wave = tid >> 5;
  const int lane = tid & 31;
  const int mloc = lane & 15;
  const int half = lane >> 4;
  const int m0   = blockIdx.y * 64;
  const int nG   = blockIdx.x * 128 + wave * 16 + mloc;

  v8f acc[4];
#pragma unroll
  for (int mt = 0; mt < 4; ++mt)
#pragma unroll
    for (int i = 0; i < 8; ++i) acc[mt][i] = 0.0f;

  const _Float16* wrowBase = W + (size_t)nG * K;
  const int am = tid >> 2;       // A-stage row (0..63)
  const int ak = (tid & 3) * 8;  // A-stage col base

  for (int k0 = 0; k0 < K; k0 += 32) {
    __syncthreads();
    if constexpr (EPI == 0) {  // f32 A: convert while staging
      const float* ap = (const float*)Av + (size_t)(m0 + am) * K + k0 + ak;
      float4 f0 = *(const float4*)(ap);
      float4 f1 = *(const float4*)(ap + 4);
      *(uint4*)(&Alds[am * 32 + ak]) = pack8(f0, f1);
    } else {  // f16 A: async global -> LDS copy (16B per thread), ASYNCcnt
      const _Float16* ap = (const _Float16*)Av + (size_t)(m0 + am) * K + k0 + ak;
      unsigned ldsAddr = (unsigned)(size_t)(&Alds[am * 32 + ak]);
      asm volatile("global_load_async_to_lds_b128 %0, %1, off"
                   :: "v"(ldsAddr), "v"(ap) : "memory");
      asm volatile("s_wait_asynccnt 0" ::: "memory");
    }
    __syncthreads();

    const _Float16* wr = wrowBase + k0;  // B[k][n] = W[n][k]
    Frag16 fb;
    fb.u[0] = *(const uint4*)(wr + half * 8);
    fb.u[1] = *(const uint4*)(wr + 16 + half * 8);

#pragma unroll
    for (int mt = 0; mt < 4; ++mt) {
      Frag16 fa;
      const int row = mt * 16 + mloc;
      fa.u[0] = *(const uint4*)(&Alds[row * 32 + half * 8]);
      fa.u[1] = *(const uint4*)(&Alds[row * 32 + 16 + half * 8]);
      acc[mt] = wmma16(fa, fb, acc[mt]);
    }
  }

  const float bv = bias[nG];
  if constexpr (EPI == 0) {
    int which = nG >> 11;
    int cc    = nG & 2047;
    int hh    = cc >> 7;
    int d     = cc & 127;
    int bb    = m0 >> 11;
    int t0    = m0 & 2047;
    size_t base = ((size_t)(bb * Hn + hh)) * Tt * HDd + d;
    if (which == 0) {  // q: store f16, pre-scaled by 1/sqrt(HD)
      const float qscale = 0.08838834764831845f;
      _Float16* qdst = (_Float16*)out0;
#pragma unroll
      for (int mt = 0; mt < 4; ++mt)
#pragma unroll
        for (int r = 0; r < 8; ++r) {
          int t = t0 + mt * 16 + r + 8 * half;
          qdst[base + (size_t)t * HDd] = (_Float16)((acc[mt][r] + bv) * qscale);
        }
    } else {           // k, v: f32 outputs in head layout
      float* dst = (which == 1) ? out1 : out2;
#pragma unroll
      for (int mt = 0; mt < 4; ++mt)
#pragma unroll
        for (int r = 0; r < 8; ++r) {
          int t = t0 + mt * 16 + r + 8 * half;
          dst[base + (size_t)t * HDd] = acc[mt][r] + bv;
        }
    }
  } else {
    float* dst = (float*)out0;
#pragma unroll
    for (int mt = 0; mt < 4; ++mt)
#pragma unroll
      for (int r = 0; r < 8; ++r) {
        size_t row = (size_t)(m0 + mt * 16 + r + 8 * half);
        dst[row * Cc + nG] = acc[mt][r] + bv;
      }
  }
}

// ---------------------------------------------------------------------------
// Flash-style attention: per (b,h), O = softmax(Q K^T / sqrt(128)) V
// grid.x = T/128, grid.y = B*H, block = 256 (8 waves x 16 query rows).
// q already f16 & pre-scaled; ctx written as f16 for the f16 GEMM3 A-path.
// ---------------------------------------------------------------------------
__global__ __launch_bounds__(256) void attn_wmma_kernel(
    const _Float16* __restrict__ qbuf, const float* __restrict__ kbuf,
    const float* __restrict__ vbuf, _Float16* __restrict__ ctx) {
  __shared__ _Float16 Klds[32 * 128];     // [key][d]  f16
  __shared__ _Float16 Vtlds[128 * 32];    // [d][key]  f16 (transposed)
  __shared__ _Float16 Plds[8 * 16 * 32];  // per-wave [m][k] probabilities

  const int tid    = threadIdx.x;
  const int wave   = tid >> 5;
  const int lane   = tid & 31;
  const int mloc   = lane & 15;
  const int half   = lane >> 4;
  const int bh     = blockIdx.y;
  const int qblock = blockIdx.x * 128;

  const _Float16* qh = qbuf + (size_t)bh * Tt * HDd;
  const float*    kh = kbuf + (size_t)bh * Tt * HDd;
  const float*    vh = vbuf + (size_t)bh * Tt * HDd;

  // Preload Q fragments (16 rows x 128 dims = 4 chunks of K=32), direct f16
  Frag16 fq[4];
  {
    const int qrow      = qblock + wave * 16 + mloc;
    const _Float16* qp  = qh + (size_t)qrow * HDd;
#pragma unroll
    for (int kc = 0; kc < 4; ++kc) {
      const _Float16* p = qp + kc * 32;
      fq[kc].u[0] = *(const uint4*)(p + half * 8);
      fq[kc].u[1] = *(const uint4*)(p + 16 + half * 8);
    }
  }

  float rowM[8], rowL[8];
  v8f Oacc[8];  // 8 chunks of 16 head dims = full 128-wide output
#pragma unroll
  for (int r = 0; r < 8; ++r) { rowM[r] = -1e30f; rowL[r] = 0.0f; }
#pragma unroll
  for (int c2 = 0; c2 < 8; ++c2)
#pragma unroll
    for (int r = 0; r < 8; ++r) Oacc[c2][r] = 0.0f;

  const int kr = tid >> 3;        // kv row this thread stages (0..31)
  const int cb = (tid & 7) * 16;  // column base (0..112)

  for (int j = 0; j < Tt; j += 32) {
    __syncthreads();
    {  // stage K tile (row-major) and V tile (transposed) as f16
      const float* src = kh + (size_t)(j + kr) * HDd + cb;
#pragma unroll
      for (int ii = 0; ii < 16; ii += 8) {
        float4 a = *(const float4*)(src + ii);
        float4 b = *(const float4*)(src + ii + 4);
        *(uint4*)(&Klds[kr * 128 + cb + ii]) = pack8(a, b);
      }
      const float* vs = vh + (size_t)(j + kr) * HDd + cb;
#pragma unroll
      for (int ii = 0; ii < 16; ii += 4) {
        float4 a = *(const float4*)(vs + ii);
        Vtlds[(cb + ii + 0) * 32 + kr] = (_Float16)a.x;
        Vtlds[(cb + ii + 1) * 32 + kr] = (_Float16)a.y;
        Vtlds[(cb + ii + 2) * 32 + kr] = (_Float16)a.z;
        Vtlds[(cb + ii + 3) * 32 + kr] = (_Float16)a.w;
      }
    }
    __syncthreads();

    // S = Q * K^T over 32 keys (two 16x16 tiles), K-dim = 128 in 4 steps
    v8f s0, s1;
#pragma unroll
    for (int i = 0; i < 8; ++i) { s0[i] = 0.0f; s1[i] = 0.0f; }
#pragma unroll
    for (int kc = 0; kc < 4; ++kc) {
      Frag16 fk;
      const _Float16* kp0 = &Klds[mloc * 128 + kc * 32];
      fk.u[0] = *(const uint4*)(kp0 + half * 8);
      fk.u[1] = *(const uint4*)(kp0 + 16 + half * 8);
      s0 = wmma16(fq[kc], fk, s0);
      const _Float16* kp1 = &Klds[(16 + mloc) * 128 + kc * 32];
      fk.u[0] = *(const uint4*)(kp1 + half * 8);
      fk.u[1] = *(const uint4*)(kp1 + 16 + half * 8);
      s1 = wmma16(fq[kc], fk, s1);
    }

    // Online softmax update; row reductions via DPP16 (no LDS traffic)
    float p0[8], p1[8];
#pragma unroll
    for (int r = 0; r < 8; ++r) {
      float a = s0[r];
      float b = s1[r];
      float mx    = rowmax16(fmaxf(a, b));
      float mnew  = fmaxf(rowM[r], mx);
      float alpha = __expf(rowM[r] - mnew);
      p0[r] = __expf(a - mnew);
      p1[r] = __expf(b - mnew);
      float sum = rowsum16(p0[r] + p1[r]);
      rowL[r] = rowL[r] * alpha + sum;
      rowM[r] = mnew;
#pragma unroll
      for (int c2 = 0; c2 < 8; ++c2) Oacc[c2][r] *= alpha;
    }

    // D-layout -> A-layout via per-wave LDS bounce
    _Float16* pw = &Plds[wave * 16 * 32];
#pragma unroll
    for (int r = 0; r < 8; ++r) {
      int mm = r + 8 * half;
      pw[mm * 32 + mloc]      = (_Float16)p0[r];
      pw[mm * 32 + 16 + mloc] = (_Float16)p1[r];
    }
    asm volatile("s_wait_dscnt 0" ::: "memory");

    Frag16 fp;
    fp.u[0] = *(const uint4*)(&pw[mloc * 32 + half * 8]);
    fp.u[1] = *(const uint4*)(&pw[mloc * 32 + 16 + half * 8]);
#pragma unroll
    for (int c2 = 0; c2 < 8; ++c2) {  // O += P * V over all 128 head dims
      Frag16 fv;
      const _Float16* vp = &Vtlds[(c2 * 16 + mloc) * 32];
      fv.u[0] = *(const uint4*)(vp + half * 8);
      fv.u[1] = *(const uint4*)(vp + 16 + half * 8);
      Oacc[c2] = wmma16(fp, fv, Oacc[c2]);
    }
  }

  // Normalize and write ctx (f16) in [B,T,C] layout
  const int b = bh >> 4;
  const int h = bh & 15;
#pragma unroll
  for (int r = 0; r < 8; ++r) {
    float rinv = 1.0f / rowL[r];
    int t = qblock + wave * 16 + r + 8 * half;
    size_t rowbase = ((size_t)(b * Tt + t)) * Cc + h * HDd;
#pragma unroll
    for (int c2 = 0; c2 < 8; ++c2) {
      ctx[rowbase + c2 * 16 + mloc] = (_Float16)(Oacc[c2][r] * rinv);
    }
  }
}

extern "C" void kernel_launch(void* const* d_in, const int* in_sizes, int n_in,
                              void* d_out, int out_size, void* d_ws, size_t ws_size,
                              hipStream_t stream) {
  (void)in_sizes; (void)n_in; (void)out_size; (void)ws_size;
  const float* x      = (const float*)d_in[0];
  const float* w_attn = (const float*)d_in[1];
  const float* b_attn = (const float*)d_in[2];
  const float* w_proj = (const float*)d_in[3];
  const float* b_proj = (const float*)d_in[4];

  float* out = (float*)d_out;
  const size_t chunk = (size_t)2 * Tt * Cc;  // B*T*C == B*H*T*HD
  float* kout = out + chunk;                 // output 2: k [B,H,T,HD]
  float* vout = out + 2 * chunk;             // output 3: v [B,H,T,HD]

  // workspace (f16): w_attn, w_proj, q(head layout, pre-scaled), ctx [B,T,C]
  _Float16* wah  = (_Float16*)d_ws;
  _Float16* wph  = wah + (size_t)3 * Cc * Cc;
  _Float16* qh   = wph + (size_t)Cc * Cc;
  _Float16* ctxh = qh + chunk;

  // 0) one-time weight conversions (L2-resident afterwards)
  {
    int n8a = 3 * Cc * Cc / 8;
    cvt_f32_f16_kernel<<<n8a / 256, 256, 0, stream>>>(w_attn, wah, n8a);
    int n8p = Cc * Cc / 8;
    cvt_f32_f16_kernel<<<n8p / 256, 256, 0, stream>>>(w_proj, wph, n8p);
  }
  // 1) qkv = x @ w_attn^T + b_attn -> q(f16,scaled), k/v(f32) head layout
  {
    dim3 grid(3 * Cc / 128, (2 * Tt) / 64);
    gemm_wmma_kernel<0><<<grid, 256, 0, stream>>>(x, wah, b_attn,
                                                  qh, kout, vout, Cc);
  }
  // 2) flash attention per (b, h)
  {
    dim3 grid(Tt / 128, 2 * Hn);
    attn_wmma_kernel<<<grid, 256, 0, stream>>>(qh, kout, vout, ctxh);
  }
  // 3) out = ctx @ w_proj^T + b_proj
  {
    dim3 grid(Cc / 128, (2 * Tt) / 64);
    gemm_wmma_kernel<1><<<grid, 256, 0, stream>>>(ctxh, wph, b_proj,
                                                  out, nullptr, nullptr, Cc);
  }
}
